// InfoNCE_loss_69526930587881
// MI455X (gfx1250) — matrix operand order repaired and checked by reading, not compiled, not run on hardware
//
#include <hip/hip_runtime.h>
#include <hip/hip_bf16.h>

typedef __attribute__((ext_vector_type(16))) _Float16 v16h;
typedef __attribute__((ext_vector_type(8)))  float    v8f;

#define NROWS   2048
#define DDIM    512
#define KCOLS   65536
#define LOG2E   1.44269504088896340736f
#define INV_T   5.0f           // 1/0.2
#define SHIFT   5.0f           // fixed logsumexp shift: |logit| <= 1/T = 5

// ---------------------------------------------------------------------------
// Kernel 1: row norms of query & keys + positive logit l_pos (one wave / row)
// ---------------------------------------------------------------------------
__global__ __launch_bounds__(256) void norms_qk_kernel(
    const float* __restrict__ q, const float* __restrict__ k,
    float* __restrict__ invq, float* __restrict__ lpos)
{
    const int w    = threadIdx.x >> 5;
    const int lane = threadIdx.x & 31;
    const int row  = blockIdx.x * 8 + w;
    const float4* qr = (const float4*)(q + (size_t)row * DDIM);
    const float4* kr = (const float4*)(k + (size_t)row * DDIM);
    float qq = 0.f, kk = 0.f, qk = 0.f;
#pragma unroll
    for (int i = 0; i < 4; ++i) {
        float4 a = qr[lane + i * 32];
        float4 b = kr[lane + i * 32];
        qq += a.x*a.x + a.y*a.y + a.z*a.z + a.w*a.w;
        kk += b.x*b.x + b.y*b.y + b.z*b.z + b.w*b.w;
        qk += a.x*b.x + a.y*b.y + a.z*b.z + a.w*b.w;
    }
#pragma unroll
    for (int m = 16; m >= 1; m >>= 1) {
        qq += __shfl_xor(qq, m, 32);
        kk += __shfl_xor(kk, m, 32);
        qk += __shfl_xor(qk, m, 32);
    }
    if (lane == 0) {
        float qn = sqrtf(qq), kn = sqrtf(kk);
        invq[row] = 1.0f / qn;
        lpos[row] = qk / fmaxf(qn * kn, 1e-8f) * INV_T;
    }
}

// ---------------------------------------------------------------------------
// Kernel 2: row norms of queue (one wave / row)
// ---------------------------------------------------------------------------
__global__ __launch_bounds__(256) void norms_u_kernel(
    const float* __restrict__ u, float* __restrict__ invu)
{
    const int w    = threadIdx.x >> 5;
    const int lane = threadIdx.x & 31;
    const int row  = blockIdx.x * 8 + w;
    const float4* ur = (const float4*)(u + (size_t)row * DDIM);
    float uu = 0.f;
#pragma unroll
    for (int i = 0; i < 4; ++i) {
        float4 a = ur[lane + i * 32];
        uu += a.x*a.x + a.y*a.y + a.z*a.z + a.w*a.w;
    }
#pragma unroll
    for (int m = 16; m >= 1; m >>= 1) uu += __shfl_xor(uu, m, 32);
    if (lane == 0) invu[row] = 1.0f / sqrtf(uu);
}

// ---------------------------------------------------------------------------
// Kernel 3: pack query f32 -> f16 in WMMA A-operand tile layout.
// Tile = 16 rows x 32 k, 512 halves, contiguous 32B per lane.
// A layout (16-bit A 16x32): lanes 0-15: K = h<8 ? h : h+8 ; lanes 16-31: +8
// ---------------------------------------------------------------------------
__global__ __launch_bounds__(256) void packA_kernel(
    const float* __restrict__ x, _Float16* __restrict__ xp)
{
    const int o    = blockIdx.x * 256 + threadIdx.x;  // < NROWS*DDIM
    const int tile = o >> 9;
    const int tw   = o & 511;
    const int lane = tw >> 4;
    const int h    = tw & 15;
    const int rb   = tile >> 4;          // row block
    const int kb   = tile & 15;          // k block (512/32)
    const int row  = rb * 16 + (lane & 15);
    int koff = (h < 8) ? h : h + 8;
    if (lane >= 16) koff += 8;
    const int kidx = kb * 32 + koff;
    xp[o] = (_Float16)x[(size_t)row * DDIM + kidx];
}

// ---------------------------------------------------------------------------
// Kernel 4: pack queue f32 -> f16 in WMMA B-operand tile layout.
// B layout (16-bit B 32x16): lanes 0-15: K = h<8 ? h : h+8 ; lanes 16-31: +8
// ---------------------------------------------------------------------------
__global__ __launch_bounds__(256) void packB_kernel(
    const float* __restrict__ x, _Float16* __restrict__ xp)
{
    const int o    = blockIdx.x * 256 + threadIdx.x;  // < KCOLS*DDIM
    const int tile = o >> 9;
    const int tw   = o & 511;
    const int lane = tw >> 4;
    const int h    = tw & 15;
    const int cb   = tile >> 4;          // column (queue-row) block
    const int kb   = tile & 15;
    const int col  = cb * 16 + (lane & 15);
    const int kidx = kb * 32 + ((h < 8) ? h : h + 8) + ((lane < 16) ? 0 : 8);
    xp[o] = (_Float16)x[(size_t)col * DDIM + kidx];
}

// ---------------------------------------------------------------------------
// Kernel 5: WMMA GEMM + fixed-shift expsum accumulation.
// Grid: 512 blocks = 16 row-groups x 32 col-groups. 8 waves/block.
// Wave w owns rows rb=(g*8+w)*16..+15 (A resident in 128 VGPRs),
// all waves stream the same 128 column tiles (WGP$ reuse).
// ---------------------------------------------------------------------------
__global__ __launch_bounds__(256) void gemm_expsum_kernel(
    const _Float16* __restrict__ qp, const _Float16* __restrict__ up,
    const float* __restrict__ invq, const float* __restrict__ invu,
    float* __restrict__ sarr)
{
    const int w      = threadIdx.x >> 5;
    const int lane   = threadIdx.x & 31;
    const int g      = blockIdx.x >> 5;         // row group   [0,16)
    const int cg     = blockIdx.x & 31;         // col group   [0,32)
    const int rb     = g * 8 + w;               // row block   [0,128)
    const int half16 = (lane < 16) ? 0 : 8;

    // Resident A tiles: 16 k-chunks x (8 VGPRs each) = 128 VGPRs
    v16h a[16];
    const _Float16* abase = qp + (size_t)rb * 8192 + lane * 16;
#pragma unroll
    for (int kb = 0; kb < 16; ++kb)
        a[kb] = *(const v16h*)(abase + kb * 512);

    // 1/||q|| for my 8 rows (rows rb*16 + half16 + j, j=0..7)
    v8f iq = *(const v8f*)(invq + rb * 16 + half16);

    const float C2 = -SHIFT * LOG2E;
    float rs[8] = {0.f, 0.f, 0.f, 0.f, 0.f, 0.f, 0.f, 0.f};

    const int cb_beg = cg * 128, cb_end = cb_beg + 128;
    for (int cb = cb_beg; cb < cb_end; ++cb) {
        const _Float16* bbase = up + (size_t)cb * 8192 + lane * 16;
        // per-lane column scale: (1/||u||) * (1/T) * log2e
        const float fu = invu[cb * 16 + (lane & 15)] * (INV_T * LOG2E);
        v8f acc = {};
#pragma unroll
        for (int kb = 0; kb < 16; ++kb) {
            v16h b = *(const v16h*)(bbase + kb * 512);
            acc = __builtin_amdgcn_wmma_f32_16x16x32_f16(
                false, a[kb], false, b, (short)0, acc, false, false);
        }
        // exp2(logit*log2e - 5*log2e), accumulate per row
#pragma unroll
        for (int j = 0; j < 8; ++j)
            rs[j] += exp2f(fmaf(acc[j], iq[j] * fu, C2));
    }

    // reduce the 16 lanes sharing each row (masks 1,2,4,8 stay in-half)
#pragma unroll
    for (int j = 0; j < 8; ++j) {
        float v = rs[j];
        v += __shfl_xor(v, 1, 32);
        v += __shfl_xor(v, 2, 32);
        v += __shfl_xor(v, 4, 32);
        v += __shfl_xor(v, 8, 32);
        rs[j] = v;
    }
    if (lane == 0 || lane == 16) {
#pragma unroll
        for (int j = 0; j < 8; ++j)
            atomicAdd(&sarr[rb * 16 + half16 + j], rs[j]);
    }
}

// ---------------------------------------------------------------------------
// Kernel 6: finalize — lse = 5 + log(s + exp(l_pos-5)); loss = mean(lse-l_pos)
// ---------------------------------------------------------------------------
__global__ __launch_bounds__(256) void finalize_kernel(
    const float* __restrict__ sarr, const float* __restrict__ lpos,
    float* __restrict__ out)
{
    const int r = blockIdx.x * 256 + threadIdx.x;
    if (r < NROWS) {
        const float lp = lpos[r];
        const float s  = sarr[r] + exp2f((lp - SHIFT) * LOG2E);
        const float c  = SHIFT + logf(s) - lp;
        atomicAdd(out, c * (1.0f / (float)NROWS));
    }
}

// ---------------------------------------------------------------------------
extern "C" void kernel_launch(void* const* d_in, const int* in_sizes, int n_in,
                              void* d_out, int out_size, void* d_ws, size_t ws_size,
                              hipStream_t stream) {
    const float* query = (const float*)d_in[0];   // [2048, 512]
    const float* keys  = (const float*)d_in[1];   // [2048, 512]
    const float* queue = (const float*)d_in[2];   // [65536, 512]
    float* out = (float*)d_out;

    char* ws = (char*)d_ws;
    _Float16* qp  = (_Float16*)(ws);                        //  2 MB packed A
    _Float16* up  = (_Float16*)(ws + (2ull  << 20));        // 64 MB packed B
    float*    invq = (float*)(ws + (66ull << 20));          // 2048
    float*    invu = invq + NROWS;                          // 65536
    float*    lpos = invu + KCOLS;                          // 2048
    float*    sarr = lpos + NROWS;                          // 2048 expsum acc

    hipMemsetAsync(out, 0, sizeof(float), stream);
    hipMemsetAsync(sarr, 0, NROWS * sizeof(float), stream);

    norms_qk_kernel<<<NROWS / 8, 256, 0, stream>>>(query, keys, invq, lpos);
    norms_u_kernel <<<KCOLS / 8, 256, 0, stream>>>(queue, invu);
    packA_kernel<<<(NROWS * DDIM) / 256, 256, 0, stream>>>(query, qp);
    packB_kernel<<<(KCOLS * DDIM) / 256, 256, 0, stream>>>(queue, up);
    gemm_expsum_kernel<<<512, 256, 0, stream>>>(qp, up, invq, invu, sarr);
    finalize_kernel<<<NROWS / 256, 256, 0, stream>>>(sarr, lpos, out);
}
